// TitansL2_60902636257289
// MI455X (gfx1250) — matrix operand order repaired and checked by reading, not compiled, not run on hardware
//
#include <hip/hip_runtime.h>
#include <hip/hip_bf16.h>

typedef __attribute__((ext_vector_type(16))) __bf16 v16bf;
typedef __attribute__((ext_vector_type(8)))  __bf16 v8bf;
typedef __attribute__((ext_vector_type(8)))  float  v8f;

// ---------------------------------------------------------------------------
// Fragment builders per CDNA5 16-bit WMMA layouts (CDNA5 ISA):
// A 16x32: lane half 0 -> K {0..7, 16..23}; half 1 -> K {8..15, 24..31}
// B 32x16: lane half 0 -> K {0..7, 16..23}; half 1 -> K {8..15, 24..31}
// ---------------------------------------------------------------------------
__device__ __forceinline__ v16bf frag_a(const __bf16* p, int half) {
  v8bf lo = *(const v8bf*)(p + half * 8);
  v8bf hi = *(const v8bf*)(p + 16 + half * 8);
  v16bf f;
#pragma unroll
  for (int i = 0; i < 8; ++i) { f[i] = lo[i]; f[i + 8] = hi[i]; }
  return f;
}

__device__ __forceinline__ v16bf frag_b(const __bf16* p, int half) {
  v8bf lo = *(const v8bf*)(p + half * 8);
  v8bf hi = *(const v8bf*)(p + 16 + half * 8);
  v16bf f;
#pragma unroll
  for (int i = 0; i < 8; ++i) { f[i] = lo[i]; f[i + 8] = hi[i]; }
  return f;
}

// ---------------------------------------------------------------------------
// float -> bf16 cast
// ---------------------------------------------------------------------------
__global__ void cast_f32_to_bf16(const float* __restrict__ in,
                                 __bf16* __restrict__ out, size_t n) {
  size_t i = (size_t)blockIdx.x * blockDim.x + threadIdx.x;
  size_t stride = (size_t)gridDim.x * blockDim.x;
  for (; i < n; i += stride) out[i] = (__bf16)in[i];
}

// ---------------------------------------------------------------------------
// C[M,N] = A[M,K] @ B[N,K]^T   (both operands row-major along K, bf16)
// One wave per block, 64x64 output tile, f32 accumulate.
// mode 0: C row-major (M x N).  mode 1: write (B,H,T,DH) head layout.
// ---------------------------------------------------------------------------
__global__ __launch_bounds__(32)
void gemm_bf16_64x64(const __bf16* __restrict__ A, const __bf16* __restrict__ Bm,
                     float* __restrict__ C, int Mdim, int Ndim, int Kdim, int mode) {
  const int lane = threadIdx.x & 31;
  const int half = lane >> 4;
  const int r    = lane & 15;
  const int m0 = blockIdx.y * 64;
  const int n0 = blockIdx.x * 64;

  v8f acc[4][4];
#pragma unroll
  for (int i = 0; i < 4; ++i)
#pragma unroll
    for (int j = 0; j < 4; ++j)
      acc[i][j] = (v8f){0.f,0.f,0.f,0.f,0.f,0.f,0.f,0.f};

  for (int kk = 0; kk < Kdim; kk += 32) {
    v16bf af[4], bf[4];
#pragma unroll
    for (int i = 0; i < 4; ++i)
      af[i] = frag_a(A + (size_t)(m0 + i * 16 + r) * Kdim + kk, half);
#pragma unroll
    for (int j = 0; j < 4; ++j)
      bf[j] = frag_b(Bm + (size_t)(n0 + j * 16 + r) * Kdim + kk, half);
#pragma unroll
    for (int i = 0; i < 4; ++i)
#pragma unroll
      for (int j = 0; j < 4; ++j)
        acc[i][j] = __builtin_amdgcn_wmma_f32_16x16x32_bf16(
            false, af[i], false, bf[j], (short)0, acc[i][j], false, false);
  }

#pragma unroll
  for (int i = 0; i < 4; ++i) {
#pragma unroll
    for (int j = 0; j < 4; ++j) {
      const int ncol = n0 + j * 16 + r;
#pragma unroll
      for (int v = 0; v < 8; ++v) {
        const int mrow = m0 + i * 16 + v + 8 * half;
        const float val = acc[i][j][v];
        size_t idx;
        if (mode == 0) {
          idx = (size_t)mrow * Ndim + ncol;
        } else {
          const int b = mrow >> 12, t = mrow & 4095;   // T = 4096
          const int h = ncol >> 6,  dh = ncol & 63;    // DH = 64
          idx = ((((size_t)b * 16 + h) * 4096) + t) * 64 + dh;
        }
        C[idx] = val;
      }
    }
  }
}

// ---------------------------------------------------------------------------
// k <- k / max(||k||_2, 1e-12)
// ---------------------------------------------------------------------------
__global__ void normalize_k(float* __restrict__ k, int rows) {
  const int rr = blockIdx.x * blockDim.x + threadIdx.x;
  if (rr >= rows) return;
  float* p = k + (size_t)rr * 64;
  float ss = 0.f;
#pragma unroll
  for (int i = 0; i < 64; ++i) ss += p[i] * p[i];
  const float inv = 1.f / fmaxf(sqrtf(ss), 1e-12f);
#pragma unroll
  for (int i = 0; i < 64; ++i) p[i] *= inv;
}

// ---------------------------------------------------------------------------
// Fully parallel intra-chunk softmax attention: y = softmax(mask(q k^T * s)) v
// grid (chunk n, bh) = (64, 64); no dependence on the memory matrix M.
// v tile staged to LDS via gfx1250 async-to-LDS DMA (ASYNCcnt).
// ---------------------------------------------------------------------------
__global__ __launch_bounds__(256)
void attn_chunks(const float* __restrict__ qf, const float* __restrict__ kf,
                 const float* __restrict__ vf, float* __restrict__ y) {
  __shared__ float  vstage[64][64];  // async-staged fp32 v chunk
  __shared__ float  sc[64][65];      // scores (padded)
  __shared__ __bf16 qb[64][64];
  __shared__ __bf16 kb[64][64];
  __shared__ __bf16 vtb[64][64];     // v^T  -> B-frag for attn@v
  __shared__ __bf16 ab[64][64];      // probs

  const int tid = threadIdx.x;
  const int n  = blockIdx.x;         // chunk
  const int bh = blockIdx.y;
  const int b = bh >> 4, h = bh & 15;
  const float scale = 0.125f;        // 1/sqrt(64)

  const size_t base = (size_t)bh * 4096 * 64 + (size_t)n * 64 * 64;

  // ---- async DMA: stage 16KB of v (fp32) into LDS ----
  {
    const unsigned ldsv = (unsigned)(uintptr_t)&vstage[0][0];
    const unsigned long long gv = (unsigned long long)(uintptr_t)(vf + base);
#pragma unroll
    for (int j = 0; j < 4; ++j) {
      const unsigned l16 = (unsigned)tid + (unsigned)j * 256u;  // 16B unit
      const unsigned dst = ldsv + l16 * 16u;
      const unsigned long long src = gv + (unsigned long long)l16 * 16u;
      asm volatile("global_load_async_to_lds_b128 %0, %1, off"
                   :: "v"(dst), "v"(src) : "memory");
    }
  }
  // ---- direct loads for q, k while the DMA flies ----
  for (int idx = tid; idx < 4096; idx += 256) {
    const int i = idx >> 6, d = idx & 63;
    qb[i][d] = (__bf16)qf[base + idx];
    kb[i][d] = (__bf16)kf[base + idx];
  }
  asm volatile("s_wait_asynccnt 0x0" ::: "memory");
  __syncthreads();

  // ---- transpose v -> bf16 ----
  for (int idx = tid; idx < 4096; idx += 256) {
    const int i = idx >> 6, d = idx & 63;
    vtb[d][i] = (__bf16)vstage[i][d];
  }
  __syncthreads();

  const int lane = tid & 31, wave = tid >> 5;
  const int half = lane >> 4, r = lane & 15;

  // ---- scores ----
#pragma unroll
  for (int tt = 0; tt < 2; ++tt) {
    const int tile = wave * 2 + tt;
    const int r0 = (tile >> 2) * 16, c0 = (tile & 3) * 16;
    v8f acc = (v8f){0.f,0.f,0.f,0.f,0.f,0.f,0.f,0.f};
#pragma unroll
    for (int kk = 0; kk < 64; kk += 32) {
      v16bf aq = frag_a(&qb[r0 + r][kk], half);
      v16bf bk = frag_b(&kb[c0 + r][kk], half);
      acc = __builtin_amdgcn_wmma_f32_16x16x32_bf16(
          false, aq, false, bk, (short)0, acc, false, false);
    }
#pragma unroll
    for (int v = 0; v < 8; ++v) {
      const int mrow = r0 + v + 8 * half;
      const int ncol = c0 + r;
      float s = acc[v] * scale;
      if (ncol > mrow) s = -__builtin_inff();
      sc[mrow][ncol] = s;
    }
  }
  __syncthreads();

  // ---- softmax ----
  if (tid < 64) {
    float mx = -__builtin_inff();
    for (int j = 0; j < 64; ++j) mx = fmaxf(mx, sc[tid][j]);
    float sum = 0.f;
    for (int j = 0; j < 64; ++j) sum += __expf(sc[tid][j] - mx);
    const float inv = 1.f / sum;
    for (int j = 0; j < 64; ++j)
      ab[tid][j] = (__bf16)(__expf(sc[tid][j] - mx) * inv);
  }
  __syncthreads();

  // ---- attn @ v -> y ----
#pragma unroll
  for (int tt = 0; tt < 2; ++tt) {
    const int tile = wave * 2 + tt;
    const int r0 = (tile >> 2) * 16, c0 = (tile & 3) * 16;
    v8f a1 = (v8f){0.f,0.f,0.f,0.f,0.f,0.f,0.f,0.f};
#pragma unroll
    for (int kk = 0; kk < 64; kk += 32) {
      v16bf afr = frag_a(&ab[r0 + r][kk], half);
      v16bf bfr = frag_b(&vtb[c0 + r][kk], half);
      a1 = __builtin_amdgcn_wmma_f32_16x16x32_bf16(
          false, afr, false, bfr, (short)0, a1, false, false);
    }
#pragma unroll
    for (int v = 0; v < 8; ++v) {
      const int i = r0 + v + 8 * half;
      const int d = c0 + r;
      const int tg = n * 64 + i;
      y[((size_t)(b * 4096 + tg)) * 1024 + h * 64 + d] = a1[v];
    }
  }
}

// ---------------------------------------------------------------------------
// Sequential part: delta-rule memory scan via exact WY transform +
// the 0.1*(q@M) memory read-out (added into y).
//   (I + alpha*L) W = beta*V - alpha*K M^T,  L[t,s] = G[s,t] (G = K K^T)
//   M_new = M + W^T K
// One workgroup per (b,h); M lives in LDS across all 64 chunks.
// ---------------------------------------------------------------------------
__global__ __launch_bounds__(256)
void chunk_scan(const float* __restrict__ qf, const float* __restrict__ kf,
                const float* __restrict__ vf,
                const float* __restrict__ alpha_raw,
                const float* __restrict__ beta_raw,
                float* __restrict__ y /* (B,T,D), holds attn_out */) {
  __shared__ float  Msh[64][64];     // memory matrix (persistent)
  __shared__ float  Gf[64][65];      // Gram K K^T (padded)
  __shared__ float  vc[64][64];      // chunk v
  __shared__ float  RHS[64][64];     // rhs, solved in-place into W
  __shared__ __bf16 qb[64][64];
  __shared__ __bf16 kb[64][64];
  __shared__ __bf16 KTb[64][64];     // k^T -> B-frag for M += W^T K
  __shared__ __bf16 Mb[64][64];      // M   -> B-frag for U0 = K M^T
  __shared__ __bf16 MTb[64][64];     // M^T -> B-frag for q@M
  __shared__ __bf16 Gt[64][64];      // Gram bf16 -> A-frag cross-block
  __shared__ __bf16 WTb[64][64];     // W^T bf16 [d][s]; zero until solved

  const int tid = threadIdx.x;
  const int bh = blockIdx.x;
  const int b = bh >> 4, h = bh & 15;

  const float alpha = 0.5f / (1.f + __expf(-alpha_raw[h]));
  const float beta  = 2.0f / (1.f + __expf(-beta_raw[h]));

  const size_t base = (size_t)bh * 4096 * 64;

  for (int idx = tid; idx < 4096; idx += 256) ((float*)Msh)[idx] = 0.f;
  __syncthreads();

  const int lane = tid & 31, wave = tid >> 5;
  const int half = lane >> 4, r = lane & 15;

  for (int n = 0; n < 64; ++n) {
    // ---- P0: stage chunk + snapshot M, zero WTb ----
    for (int idx = tid; idx < 4096; idx += 256) {
      const int i = idx >> 6, d = idx & 63;
      const size_t g = base + (size_t)(n * 64 + i) * 64 + d;
      const float qv = qf[g], kv = kf[g], vv = vf[g];
      qb[i][d]  = (__bf16)qv;
      kb[i][d]  = (__bf16)kv;
      KTb[d][i] = (__bf16)kv;
      vc[i][d]  = vv;
      const float m = Msh[i][d];
      Mb[i][d]  = (__bf16)m;
      MTb[d][i] = (__bf16)m;
      WTb[i][d] = (__bf16)0.f;
    }
    __syncthreads();

    // ---- P1: Gram G = k k^T (symmetric) ----
#pragma unroll
    for (int tt = 0; tt < 2; ++tt) {
      const int tile = wave * 2 + tt;
      const int r0 = (tile >> 2) * 16, c0 = (tile & 3) * 16;
      v8f accG = (v8f){0.f,0.f,0.f,0.f,0.f,0.f,0.f,0.f};
#pragma unroll
      for (int kk = 0; kk < 64; kk += 32) {
        v16bf ak = frag_a(&kb[r0 + r][kk], half);
        v16bf bk = frag_b(&kb[c0 + r][kk], half);
        accG = __builtin_amdgcn_wmma_f32_16x16x32_bf16(
            false, ak, false, bk, (short)0, accG, false, false);
      }
#pragma unroll
      for (int v = 0; v < 8; ++v) {
        const int mrow = r0 + v + 8 * half;
        const int ncol = c0 + r;
        const float g = accG[v];
        Gf[mrow][ncol] = g;
        Gt[mrow][ncol] = (__bf16)g;
      }
    }
    __syncthreads();

    // ---- P3: y += 0.1*(q@M); RHS = beta*V - alpha*(K M^T) ----
#pragma unroll
    for (int tt = 0; tt < 2; ++tt) {
      const int tile = wave * 2 + tt;
      const int r0 = (tile >> 2) * 16, c0 = (tile & 3) * 16;
      v8f a2 = (v8f){0.f,0.f,0.f,0.f,0.f,0.f,0.f,0.f};
      v8f aU = a2;
#pragma unroll
      for (int kk = 0; kk < 64; kk += 32) {
        v16bf afq = frag_a(&qb[r0 + r][kk], half);
        v16bf bfm = frag_b(&MTb[c0 + r][kk], half);
        a2 = __builtin_amdgcn_wmma_f32_16x16x32_bf16(
            false, afq, false, bfm, (short)0, a2, false, false);
        v16bf afk = frag_a(&kb[r0 + r][kk], half);
        v16bf bmm = frag_b(&Mb[c0 + r][kk], half);
        aU = __builtin_amdgcn_wmma_f32_16x16x32_bf16(
            false, afk, false, bmm, (short)0, aU, false, false);
      }
#pragma unroll
      for (int v = 0; v < 8; ++v) {
        const int i = r0 + v + 8 * half;
        const int d = c0 + r;
        const int tg = n * 64 + i;
        const size_t yi = ((size_t)(b * 4096 + tg)) * 1024 + h * 64 + d;
        y[yi] += 0.1f * a2[v];
        RHS[i][d] = beta * vc[i][d] - alpha * aU[v];
      }
    }
    __syncthreads();

    // ---- P4: solve (I + alpha*L) W = RHS, blocks of 16 ----
    for (int jb = 0; jb < 4; ++jb) {
      const int bs = jb * 16;
      if (jb > 0) {
        // cross-block update (WTb columns s >= bs are still zero)
        if (wave < 4) {
          const int c0 = wave * 16;
          v8f acc = (v8f){0.f,0.f,0.f,0.f,0.f,0.f,0.f,0.f};
#pragma unroll
          for (int kk = 0; kk < 64; kk += 32) {
            v16bf ag = frag_a(&Gt[bs + r][kk], half);
            v16bf bw = frag_b(&WTb[c0 + r][kk], half);
            acc = __builtin_amdgcn_wmma_f32_16x16x32_bf16(
                false, ag, false, bw, (short)0, acc, false, false);
          }
#pragma unroll
          for (int v = 0; v < 8; ++v)
            RHS[bs + v + 8 * half][c0 + r] -= alpha * acc[v];
        }
        __syncthreads();
      }
      // diagonal 16x16 triangular solve: waves 0 and 1 (one d per lane)
      if (wave < 2) {
        float w[16];
        const int d0 = wave * 32 + lane;
#pragma unroll
        for (int t = 0; t < 16; ++t) {
          float a0 = RHS[bs + t][d0];
#pragma unroll
          for (int s = 0; s < t; ++s)
            a0 -= alpha * Gf[bs + s][bs + t] * w[s];   // alpha*L[t,s]
          w[t] = a0;
          RHS[bs + t][d0] = a0;
          WTb[d0][bs + t] = (__bf16)a0;
        }
      }
      __syncthreads();
    }

    // ---- P5: M += W^T K ----
#pragma unroll
    for (int tt = 0; tt < 2; ++tt) {
      const int tile = wave * 2 + tt;
      const int r0 = (tile >> 2) * 16, c0 = (tile & 3) * 16;
      v8f acc = (v8f){0.f,0.f,0.f,0.f,0.f,0.f,0.f,0.f};
#pragma unroll
      for (int kk = 0; kk < 64; kk += 32) {
        v16bf aw = frag_a(&WTb[r0 + r][kk], half);
        v16bf bk = frag_b(&KTb[c0 + r][kk], half);
        acc = __builtin_amdgcn_wmma_f32_16x16x32_bf16(
            false, aw, false, bk, (short)0, acc, false, false);
      }
#pragma unroll
      for (int v = 0; v < 8; ++v)
        Msh[r0 + v + 8 * half][c0 + r] += acc[v];
    }
    __syncthreads();
  }
}

// ---------------------------------------------------------------------------
extern "C" void kernel_launch(void* const* d_in, const int* in_sizes, int n_in,
                              void* d_out, int out_size, void* d_ws, size_t ws_size,
                              hipStream_t stream) {
  const float* x   = (const float*)d_in[0];
  const float* Wq  = (const float*)d_in[1];
  const float* Wk  = (const float*)d_in[2];
  const float* Wv  = (const float*)d_in[3];
  const float* Wo  = (const float*)d_in[4];
  const float* ar  = (const float*)d_in[5];
  const float* br  = (const float*)d_in[6];
  float* out = (float*)d_out;

  const size_t NX = (size_t)4 * 4096 * 1024;   // B*T*D
  const size_t NW = (size_t)1024 * 1024;

  char* ws = (char*)d_ws;
  size_t off = 0;
  __bf16* xb  = (__bf16*)(ws + off); off += NX * 2;
  __bf16* Wqb = (__bf16*)(ws + off); off += NW * 2;
  __bf16* Wkb = (__bf16*)(ws + off); off += NW * 2;
  __bf16* Wvb = (__bf16*)(ws + off); off += NW * 2;
  __bf16* Wob = (__bf16*)(ws + off); off += NW * 2;
  float*  qf  = (float*)(ws + off);  off += NX * 4;
  float*  kf  = (float*)(ws + off);  off += NX * 4;
  float*  vf  = (float*)(ws + off);  off += NX * 4;
  float*  yf  = (float*)(ws + off);  off += NX * 4;
  __bf16* yb  = (__bf16*)(ws + off); off += NX * 2;

  // 1) bf16 casts of x and weights
  cast_f32_to_bf16<<<4096, 256, 0, stream>>>(x,  xb,  NX);
  cast_f32_to_bf16<<<1024, 256, 0, stream>>>(Wq, Wqb, NW);
  cast_f32_to_bf16<<<1024, 256, 0, stream>>>(Wk, Wkb, NW);
  cast_f32_to_bf16<<<1024, 256, 0, stream>>>(Wv, Wvb, NW);
  cast_f32_to_bf16<<<1024, 256, 0, stream>>>(Wo, Wob, NW);

  // 2) Q/K/V projections -> (B,H,T,DH)
  dim3 ggrid(1024 / 64, 16384 / 64);
  gemm_bf16_64x64<<<ggrid, 32, 0, stream>>>(xb, Wqb, qf, 16384, 1024, 1024, 1);
  gemm_bf16_64x64<<<ggrid, 32, 0, stream>>>(xb, Wkb, kf, 16384, 1024, 1024, 1);
  gemm_bf16_64x64<<<ggrid, 32, 0, stream>>>(xb, Wvb, vf, 16384, 1024, 1024, 1);

  // 3) L2-normalize k along DH
  normalize_k<<<(262144 + 255) / 256, 256, 0, stream>>>(kf, 262144);

  // 4a) parallel intra-chunk attention -> yf = attn_out (4096 WGs)
  attn_chunks<<<dim3(64, 64), 256, 0, stream>>>(qf, kf, vf, yf);

  // 4b) sequential WY-transform delta-rule scan; yf += 0.1 * q@M (64 WGs)
  chunk_scan<<<64, 256, 0, stream>>>(qf, kf, vf, ar, br, yf);

  // 5) output projection: yf @ Wo^T -> out
  cast_f32_to_bf16<<<4096, 256, 0, stream>>>(yf, yb, NX);
  gemm_bf16_64x64<<<ggrid, 32, 0, stream>>>(yb, Wob, out, 16384, 1024, 1024, 0);
}
